// DCN_32298154066646
// MI455X (gfx1250) — matrix operand order repaired and checked by reading, not compiled, not run on hardware
//
#include <hip/hip_runtime.h>

#define K2      9
#define CIN     256
#define COUT    256
#define HIMG    64
#define WIMG    64
#define BATCH   4
#define HW      (HIMG * WIMG)          // 4096
#define KDIM    (CIN * K2)             // 2304
#define NCHUNK  (KDIM / 32)            // 72
#define BN_EPS  1e-5f

typedef __attribute__((ext_vector_type(16))) _Float16     v16h;
typedef __attribute__((ext_vector_type(8)))  _Float16     v8h;
typedef __attribute__((ext_vector_type(8)))  float        v8f;
typedef __attribute__((ext_vector_type(4)))  int          i32x4;
typedef __attribute__((ext_vector_type(4)))  float        f32x4;
typedef __attribute__((ext_vector_type(4)))  unsigned int u32x4;

// ---- CDNA5 async global->LDS path (probe-guarded; sync fallback otherwise) ----
#if defined(__AMDGCN__) && __has_builtin(__builtin_amdgcn_global_load_async_to_lds_b128)
#define USE_ASYNC_LDS 1
typedef __attribute__((address_space(1))) i32x4 gi32x4;   // global int4 (AS1)
typedef __attribute__((address_space(3))) i32x4 li32x4;   // LDS    int4 (AS3)
#define GPTR(p) ((gi32x4*)(unsigned long long)(p))
#define LPTR(p) ((li32x4*)(unsigned)(unsigned long long)(p))
#if __has_builtin(__builtin_amdgcn_s_wait_asynccnt)
#define WAIT_ASYNC0() __builtin_amdgcn_s_wait_asynccnt(0)
#else
#define WAIT_ASYNC0() asm volatile("s_wait_asynccnt 0x0" ::: "memory")
#endif
#else
#define USE_ASYNC_LDS 0
#define WAIT_ASYNC0()
#endif

// --------------------------------------------------------------------------
// Kernel 1: offset/mask 3x3 conv (27 ch) + fold bilinear sampling into
// 4 corner plane-offsets + 4 weights (mask & validity pre-multiplied).
// One thread per (b, kcell, ho, wo). Block = one row (64 threads, 2 waves).
// --------------------------------------------------------------------------
__global__ __launch_bounds__(64) void dcn_prep(
    const float* __restrict__ x,      const float* __restrict__ w_off,
    const float* __restrict__ b_off,  const float* __restrict__ w_mask,
    const float* __restrict__ b_mask, i32x4* __restrict__ goff,
    f32x4* __restrict__ gwgt)
{
    const int blk   = blockIdx.x;
    const int ho    = blk % HIMG;
    const int kcell = (blk / HIMG) % K2;
    const int b     = blk / (HIMG * K2);
    const int wo    = threadIdx.x;

    const float* xb = x      + (size_t)b * CIN * HW;
    const float* wy = w_off  + (size_t)(2 * kcell)     * CIN * K2;
    const float* wx = w_off  + (size_t)(2 * kcell + 1) * CIN * K2;
    const float* wm = w_mask + (size_t)kcell           * CIN * K2;

    float ay = 0.f, ax = 0.f, am = 0.f;
    for (int c = 0; c < CIN; ++c) {
        const float* xc  = xb + c * HW;
        const float* wyc = wy + c * K2;
        const float* wxc = wx + c * K2;
        const float* wmc = wm + c * K2;
        #pragma unroll
        for (int ky = 0; ky < 3; ++ky) {
            const int iy = ho - 1 + ky;
            if (iy < 0 || iy >= HIMG) continue;
            #pragma unroll
            for (int kx = 0; kx < 3; ++kx) {
                const int ix = wo - 1 + kx;
                const float xv = (ix >= 0 && ix < WIMG) ? xc[iy * WIMG + ix] : 0.f;
                const int t = ky * 3 + kx;
                ay += xv * wyc[t];
                ax += xv * wxc[t];
                am += xv * wmc[t];
            }
        }
    }
    ay += b_off[2 * kcell];
    ax += b_off[2 * kcell + 1];
    am += b_mask[kcell];

    const float py = (float)(ho - 1 + kcell / 3) + ay;
    const float px = (float)(wo - 1 + kcell % 3) + ax;
    const float mv = 2.f / (1.f + __expf(-am));

    const float y0 = floorf(py), x0 = floorf(px);
    const float wy1 = py - y0, wx1 = px - x0;
    const float wyv[2] = {1.f - wy1, wy1};
    const float wxv[2] = {1.f - wx1, wx1};

    i32x4 of; f32x4 wg;
    int   otmp[4]; float wtmp[4];
    #pragma unroll
    for (int dy = 0; dy < 2; ++dy) {
        #pragma unroll
        for (int dx = 0; dx < 2; ++dx) {
            const float yf = y0 + (float)dy;
            const float xf = x0 + (float)dx;
            const bool valid = (yf >= 0.f) && (yf <= (float)(HIMG - 1)) &&
                               (xf >= 0.f) && (xf <= (float)(WIMG - 1));
            const int yi = (int)fminf(fmaxf(yf, 0.f), (float)(HIMG - 1));
            const int xi = (int)fminf(fmaxf(xf, 0.f), (float)(WIMG - 1));
            otmp[dy * 2 + dx] = yi * WIMG + xi;
            wtmp[dy * 2 + dx] = (valid ? wyv[dy] * wxv[dx] : 0.f) * mv;
        }
    }
    of.x = otmp[0]; of.y = otmp[1]; of.z = otmp[2]; of.w = otmp[3];
    wg.x = wtmp[0]; wg.y = wtmp[1]; wg.z = wtmp[2]; wg.w = wtmp[3];

    const int rec = (b * K2 + kcell) * HW + ho * WIMG + wo;
    goff[rec] = of;
    gwgt[rec] = wg;
}

// --------------------------------------------------------------------------
// Kernel 2: weight transform  w[o][c][kcell] f32  ->
//   wB[(((kcell*8 + c/32)*256 + o)*32 + c%32)] f16
// so each GEMM K-chunk's B tile is one contiguous 16KB block in [n][k] layout.
// --------------------------------------------------------------------------
__global__ __launch_bounds__(256) void dcn_wxform(
    const float* __restrict__ w, _Float16* __restrict__ wB)
{
    const int e  = blockIdx.x * 256 + threadIdx.x;   // 9*8*256*32 = 589824 total
    const int cw = e & 31;
    const int o  = (e >> 5) & 255;
    const int cb = (e >> 13) & 7;
    const int kc = e >> 16;
    const int c  = cb * 32 + cw;
    wB[e] = (_Float16)w[((size_t)o * CIN + c) * K2 + kc];
}

// --------------------------------------------------------------------------
// Kernel 3: fused deformable-sampling + GEMM (WMMA f16->f32) + bias/BN/ReLU.
// Block = 256 threads (8 waves). Tile: M=32 positions x N=256 out-channels.
// Double-buffered K-chunks: B tile streamed with GLOBAL_LOAD_ASYNC_TO_LDS_B128
// (ASYNCcnt), A tile gathered/converted by VALU — both for chunk kc+1 while
// chunk kc is consumed by 4x v_wmma_f32_16x16x32_f16 per wave.
// --------------------------------------------------------------------------
__global__ __launch_bounds__(256) void dcn_gemm(
    const float*    __restrict__ x,
    const i32x4*    __restrict__ goff,
    const f32x4*    __restrict__ gwgt,
    const _Float16* __restrict__ wB,
    const float*    __restrict__ bconv,
    const float*    __restrict__ gamma,
    const float*    __restrict__ beta,
    const float*    __restrict__ rmean,
    const float*    __restrict__ rvar,
    float*          __restrict__ out)
{
    __shared__ _Float16 Alds[2][32 * 32];    //  4 KB   A tiles [buf][m][k]
    __shared__ _Float16 Blds[2][256 * 32];   // 32 KB   B tiles [buf][n][k]
    __shared__ float    sScale[COUT];
    __shared__ float    sShift[COUT];
    __shared__ float    outStage[32 * 257];  // ~32.9 KB, pitch 257 vs bank conflicts

    const int tid  = threadIdx.x;
    const int lane = tid & 31;
    const int wv   = tid >> 5;
    const int wv_m = wv >> 2;                // 0..1
    const int wv_n = wv & 3;                 // 0..3

    const int m0     = blockIdx.x * 32;      // 32 consecutive positions: same b, same row
    const int b      = m0 >> 12;
    const int posim0 = m0 & (HW - 1);

    // fold conv-bias + batchnorm into per-channel scale/shift
    {
        const int o = tid;
        const float s = gamma[o] * rsqrtf(rvar[o] + BN_EPS);
        sScale[o] = s;
        sShift[o] = bconv[o] * s + beta[o] - rmean[o] * s;
    }

    v8f acc[4] = {};

    const float* xb   = x + (size_t)b * CIN * HW;
    const int    posA = tid & 31;            // position handled in A-fill
    const int    clA  = tid >> 5;            // channel sub-lane 0..7
    const int    recBase = b * K2 * HW + posim0 + posA;

    // ---- B fill: one contiguous 16 KB chunk, straight into LDS ----
    auto copyB = [&](int kc, int buf) {
#if USE_ASYNC_LDS
        const char* g = (const char*)wB + (size_t)kc * 16384 + tid * 64;
        const char* l = (const char*)&Blds[buf][0] + tid * 64;
        __builtin_amdgcn_global_load_async_to_lds_b128(GPTR(g), LPTR(l), 0,  0);
        __builtin_amdgcn_global_load_async_to_lds_b128(GPTR(g), LPTR(l), 16, 0);
        __builtin_amdgcn_global_load_async_to_lds_b128(GPTR(g), LPTR(l), 32, 0);
        __builtin_amdgcn_global_load_async_to_lds_b128(GPTR(g), LPTR(l), 48, 0);
#else
        const u32x4* src = (const u32x4*)(wB + (size_t)kc * 8192);
        u32x4* dst = (u32x4*)&Blds[buf][0];
        #pragma unroll
        for (int i = 0; i < 4; ++i)
            dst[tid * 4 + i] = src[tid * 4 + i];
#endif
    };

    // ---- A fill: deformable bilinear sample -> f16, 4 channels/thread ----
    auto fillA = [&](int kc, int buf) {
        const int kcell = kc >> 3;
        const int c0    = (kc & 7) * 32;
        const i32x4 of = goff[recBase + kcell * HW];
        const f32x4 wg = gwgt[recBase + kcell * HW];
        #pragma unroll
        for (int q = 0; q < 4; ++q) {
            const int kloc = clA + q * 8;
            const float* xc = xb + (size_t)(c0 + kloc) * HW;
            const float v = wg.x * xc[of.x] + wg.y * xc[of.y] +
                            wg.z * xc[of.z] + wg.w * xc[of.w];
            Alds[buf][posA * 32 + kloc] = (_Float16)v;
        }
    };

    // prologue: stage chunk 0 into buffer 0
    copyB(0, 0);
    fillA(0, 0);

    for (int kc = 0; kc < NCHUNK; ++kc) {
        const int cur = kc & 1;
        const int nxt = cur ^ 1;

        WAIT_ASYNC0();                       // my async B[kc] transfers landed in LDS
        __syncthreads();                     // everyone's A/B for kc visible

        if (kc + 1 < NCHUNK) {               // stage kc+1 while consuming kc
            copyB(kc + 1, nxt);
            fillA(kc + 1, nxt);
        }

        // ---- WMMA consume: per-ISA 16-bit A 16x32 lane layout ----
        {
            const _Float16* Ab = &Alds[cur][0];
            const _Float16* Bb = &Blds[cur][0];
            const int mrow = wv_m * 16 + (lane & 15);
            const int asel = (lane < 16) ? 0 : 8;    // K {0-7,16-23} vs {8-15,24-31}
            const v8h alo = *(const v8h*)(Ab + mrow * 32 + asel);
            const v8h ahi = *(const v8h*)(Ab + mrow * 32 + asel + 16);
            const v16h a = __builtin_shufflevector(alo, ahi,
                0, 1, 2, 3, 4, 5, 6, 7, 8, 9, 10, 11, 12, 13, 14, 15);

            const int ncol = lane & 15;
            const int bsel = (lane < 16) ? 0 : 8;    // K {0-7,16-23} vs {8-15,24-31}
            #pragma unroll
            for (int t = 0; t < 4; ++t) {
                const int n = wv_n * 64 + t * 16 + ncol;
                const v8h blo = *(const v8h*)(Bb + n * 32 + bsel);
                const v8h bhi = *(const v8h*)(Bb + n * 32 + bsel + 16);
                const v16h bm = __builtin_shufflevector(blo, bhi,
                    0, 1, 2, 3, 4, 5, 6, 7, 8, 9, 10, 11, 12, 13, 14, 15);
                acc[t] = __builtin_amdgcn_wmma_f32_16x16x32_f16(
                    false, a, false, bm, (short)0, acc[t], false, false);
            }
        }
    }

    // ---- epilogue: stage accumulators through LDS, then coalesced NCHW stores
    __syncthreads();
    {
        const int mb = wv_m * 16 + ((lane >> 4) << 3);   // C/D layout: lanes>=16 -> M+8
        #pragma unroll
        for (int t = 0; t < 4; ++t) {
            const int n = wv_n * 64 + t * 16 + (lane & 15);
            #pragma unroll
            for (int r = 0; r < 8; ++r)
                outStage[(mb + r) * 257 + n] = acc[t][r];
        }
    }
    __syncthreads();
    {
        const int pos = tid & 31;
        const int ob  = tid >> 5;
        float* outp = out + (size_t)b * COUT * HW + posim0 + pos;
        #pragma unroll 4
        for (int i = 0; i < 32; ++i) {
            const int o = ob + i * 8;
            float v = outStage[pos * 257 + o];
            v = fmaxf(v * sScale[o] + sShift[o], 0.f);
            outp[(size_t)o * HW] = v;
        }
    }
}

// --------------------------------------------------------------------------
// Host launcher. Workspace layout (needs ~5.9 MB):
//   [0)           i32x4 goff[4*9*4096]   = 2,359,296 B
//   [2359296)     f32x4 gwgt[4*9*4096]   = 2,359,296 B
//   [4718592)     f16   wB[9*8*256*32]   = 1,179,648 B
// --------------------------------------------------------------------------
extern "C" void kernel_launch(void* const* d_in, const int* in_sizes, int n_in,
                              void* d_out, int out_size, void* d_ws, size_t ws_size,
                              hipStream_t stream)
{
    const float* x      = (const float*)d_in[0];
    const float* w_off  = (const float*)d_in[1];
    const float* b_off  = (const float*)d_in[2];
    const float* w_mask = (const float*)d_in[3];
    const float* b_mask = (const float*)d_in[4];
    const float* w      = (const float*)d_in[5];
    const float* bconv  = (const float*)d_in[6];
    const float* gamma  = (const float*)d_in[7];
    const float* beta   = (const float*)d_in[8];
    const float* rmean  = (const float*)d_in[9];
    const float* rvar   = (const float*)d_in[10];
    float* out = (float*)d_out;

    char* ws = (char*)d_ws;
    i32x4*    goff = (i32x4*)ws;
    f32x4*    gwgt = (f32x4*)(ws + 2359296);
    _Float16* wB   = (_Float16*)(ws + 4718592);

    dcn_prep  <<<BATCH * K2 * HIMG, 64,  0, stream>>>(x, w_off, b_off, w_mask, b_mask,
                                                      goff, gwgt);
    dcn_wxform<<<2304,              256, 0, stream>>>(w, wB);
    dcn_gemm  <<<512,               256, 0, stream>>>(x, goff, gwgt, wB, bconv,
                                                      gamma, beta, rmean, rvar, out);
}
